// CombinedScoreNetwork_82523501625601
// MI455X (gfx1250) — compile-verified, matrix operand order checked by reading, not run on hardware
//
#include <hip/hip_runtime.h>
#include <hip/hip_bf16.h>

// CDNA5 / gfx1250 implementation of the GDSS score network.
// All heavy contractions go through v_wmma_f32_16x16x32_f16 (wave32, f32 acc).
// Weights are pre-packed on device into WMMA B-fragment order; the dominant
// 1M-row edge/final MLP GEMMs gather their A fragments directly from
// x_out / adj channels (no materialized input matrix).

typedef _Float16 half_t;
typedef __attribute__((ext_vector_type(16))) _Float16 v16h;
typedef __attribute__((ext_vector_type(8)))  _Float16 v8h;
typedef __attribute__((ext_vector_type(8)))  float    v8f;

#define NB   16      // batch
#define NN_  256     // nodes
#define NN2  65536   // N*N
#define CHT  22      // total adjacency channels accumulated
#define ME   1048576 // NB*NN2 edge rows

// ---------------------------------------------------------------------------
// Generic batched GEMM:  Out[bz] = act( A[bz] (MxK,f16) * B[bz] (KxNpad,f16) + bias )
// K mult of 32, M mult of 16, Npad mult of 16. One wave = 16 rows x (NTILE*16) cols.
// ACT: 0=none 1=ELU 2=tanh(v*rowscale[row]).  OF16: f16 output else f32.
// BPACK: B stored in fragment order [ntile][kchunk][lane][16], else row-major.
// ---------------------------------------------------------------------------
template<int ACT, int OF16, int NTILE, int BPACK>
__global__ __launch_bounds__(256) void gemm_k(
    const half_t* __restrict__ A, int lda, long sA,
    const half_t* __restrict__ Bm, int ldb, long sB,
    void* __restrict__ Out, int ldo, long sOut,
    const float* __restrict__ bias, int bias_sub, int bias_ld,
    const float* __restrict__ rowscale,
    int M, int K, int Npad)
{
  int bz = blockIdx.y;
  const half_t* Ab = A + (long)bz * sA;
  const half_t* Bb = Bm + (long)bz * sB;
  int mtiles  = M >> 4;
  int ngroups = (Npad >> 4) / NTILE;
  int wid = blockIdx.x * (blockDim.x >> 5) + (threadIdx.x >> 5);
  if (wid >= mtiles * ngroups) return;
  int tm = wid / ngroups, tg = wid % ngroups;
  int lane = threadIdx.x & 31;
  int l15  = lane & 15;
  int kb   = (lane >= 16) ? 8 : 0;       // documented A-frag K pattern
  int arow = tm * 16 + l15;
  int nch  = K >> 5;

  v8f acc[NTILE];
#pragma unroll
  for (int nt = 0; nt < NTILE; nt++) acc[nt] = {};

  for (int kc = 0; kc < K; kc += 32) {
    const half_t* ap = Ab + (long)arow * lda + kc + kb;
    v8h alo = *(const v8h*)ap;
    v8h ahi = *(const v8h*)(ap + 16);
    v16h af;
#pragma unroll
    for (int t = 0; t < 8; t++) { af[t] = alo[t]; af[t + 8] = ahi[t]; }
    int kci = kc >> 5;
#pragma unroll
    for (int nt = 0; nt < NTILE; nt++) {
      int tnIdx = tg * NTILE + nt;
      v16h bf;
      if (BPACK) {
        bf = *(const v16h*)(Bb + (((long)tnIdx * nch + kci) * 32 + lane) * 16);
      } else {
        int bcol = tnIdx * 16 + l15;
#pragma unroll
        for (int t = 0; t < 16; t++) {
          int k = kc + ((t < 8) ? (kb + t) : (16 + kb + (t - 8)));
          bf[t] = Bb[(long)k * ldb + bcol];
        }
      }
      acc[nt] = __builtin_amdgcn_wmma_f32_16x16x32_f16(
          false, af, false, bf, (short)0, acc[nt], false, false);
    }
  }

  long obase = (long)bz * sOut;
  int rbase = tm * 16 + ((lane >= 16) ? 8 : 0);   // documented C/D layout
#pragma unroll
  for (int nt = 0; nt < NTILE; nt++) {
    int bcol = (tg * NTILE + nt) * 16 + l15;
    float bv = 0.f;
    if (bias) bv = bias[(bz % bias_sub) * bias_ld + bcol];
#pragma unroll
    for (int r = 0; r < 8; r++) {
      float v = acc[nt][r] + bv;
      if (ACT == 1) v = (v > 0.f) ? v : (__expf(v) - 1.f);
      else if (ACT == 2) v = tanhf(v * (rowscale ? rowscale[rbase + r] : 1.f));
      if (OF16) ((half_t*)Out)[obase + (long)(rbase + r) * ldo + bcol] = (half_t)v;
      else      ((float*)Out)[obase + (long)(rbase + r) * ldo + bcol] = v;
    }
  }
}

// ---------------------------------------------------------------------------
// Fused edge MLP layer-1: H = ELU([x_i | x_j | adj_c] @ W1 + b1), 1M rows.
// One wave = 16 edge rows (same b,i, consecutive j) x 32 cols. K=96 (3 chunks).
// A fragments gathered on the fly; B in fragment order (nch=3, 2 n-tiles).
// ---------------------------------------------------------------------------
__global__ __launch_bounds__(256) void edge_gemm1_k(
    const half_t* __restrict__ xout, const float* __restrict__ adjlist,
    int ch_in, int ci,
    const half_t* __restrict__ Bfrag, const float* __restrict__ bias,
    half_t* __restrict__ Out)
{
  int wid = blockIdx.x * (blockDim.x >> 5) + (threadIdx.x >> 5);
  if (wid >= (ME >> 4)) return;
  int lane = threadIdx.x & 31;
  int l15  = lane & 15;
  int kb   = (lane >= 16) ? 8 : 0;
  long eb = (long)wid << 4;
  int b = (int)(eb >> 16); int r = (int)(eb & (NN2 - 1));
  int i = r >> 8; int j = (r & 255) + l15;

  v8f acc0 = {}, acc1 = {};
  v16h af, bf;

  // chunk 0: K=0..31 -> x_out[b,i,:]
  {
    const half_t* p = xout + ((b << 8) + i) * 32 + kb;
    v8h lo = *(const v8h*)p; v8h hi = *(const v8h*)(p + 16);
#pragma unroll
    for (int t = 0; t < 8; t++) { af[t] = lo[t]; af[t + 8] = hi[t]; }
    bf = *(const v16h*)(Bfrag + ((0 * 3 + 0) * 32 + lane) * 16);
    acc0 = __builtin_amdgcn_wmma_f32_16x16x32_f16(false, af, false, bf, (short)0, acc0, false, false);
    bf = *(const v16h*)(Bfrag + ((1 * 3 + 0) * 32 + lane) * 16);
    acc1 = __builtin_amdgcn_wmma_f32_16x16x32_f16(false, af, false, bf, (short)0, acc1, false, false);
  }
  // chunk 1: K=32..63 -> x_out[b,j,:]
  {
    const half_t* p = xout + ((b << 8) + j) * 32 + kb;
    v8h lo = *(const v8h*)p; v8h hi = *(const v8h*)(p + 16);
#pragma unroll
    for (int t = 0; t < 8; t++) { af[t] = lo[t]; af[t + 8] = hi[t]; }
    bf = *(const v16h*)(Bfrag + ((0 * 3 + 1) * 32 + lane) * 16);
    acc0 = __builtin_amdgcn_wmma_f32_16x16x32_f16(false, af, false, bf, (short)0, acc0, false, false);
    bf = *(const v16h*)(Bfrag + ((1 * 3 + 1) * 32 + lane) * 16);
    acc1 = __builtin_amdgcn_wmma_f32_16x16x32_f16(false, af, false, bf, (short)0, acc1, false, false);
  }
  // chunk 2: K=64..95 -> adj channels c = K-64 (c >= ci is zero pad)
  {
#pragma unroll
    for (int t = 0; t < 16; t++) af[t] = (half_t)0.f;
#pragma unroll
    for (int t = 0; t < 8; t++) {
      int c = kb + t;
      if (c < ci)
        af[t] = (half_t)adjlist[(((long)b * CHT + ch_in + c) << 16) + (i << 8) + j];
    }
    bf = *(const v16h*)(Bfrag + ((0 * 3 + 2) * 32 + lane) * 16);
    acc0 = __builtin_amdgcn_wmma_f32_16x16x32_f16(false, af, false, bf, (short)0, acc0, false, false);
    bf = *(const v16h*)(Bfrag + ((1 * 3 + 2) * 32 + lane) * 16);
    acc1 = __builtin_amdgcn_wmma_f32_16x16x32_f16(false, af, false, bf, (short)0, acc1, false, false);
  }

  long rbase = eb + ((lane >= 16) ? 8 : 0);
#pragma unroll
  for (int nt = 0; nt < 2; nt++) {
    v8f* ap = nt ? &acc1 : &acc0;
    int bcol = nt * 16 + l15;
    float bv = bias[bcol];
#pragma unroll
    for (int rr = 0; rr < 8; rr++) {
      float v = (*ap)[rr] + bv;
      v = (v > 0.f) ? v : (__expf(v) - 1.f);
      Out[(rbase + rr) * 32 + bcol] = (half_t)v;
    }
  }
}

// ---------------------------------------------------------------------------
// Fused final MLP layer-1: H = ELU(adj_channels(22,pad32) @ W1 + b1), 1M rows.
// K=32 (1 chunk), Npad=64 (4 n-tiles). A gathered from adj_list.
// ---------------------------------------------------------------------------
__global__ __launch_bounds__(256) void fin_gemm1_k(
    const float* __restrict__ adjlist,
    const half_t* __restrict__ Bfrag, const float* __restrict__ bias,
    half_t* __restrict__ Out)
{
  int wid = blockIdx.x * (blockDim.x >> 5) + (threadIdx.x >> 5);
  if (wid >= (ME >> 4)) return;
  int lane = threadIdx.x & 31;
  int l15  = lane & 15;
  int kb   = (lane >= 16) ? 8 : 0;
  long eb = (long)wid << 4;
  int b = (int)(eb >> 16); int r = (int)(eb & (NN2 - 1));
  int i = r >> 8; int j = (r & 255) + l15;

  v16h af;
#pragma unroll
  for (int t = 0; t < 16; t++) af[t] = (half_t)0.f;
#pragma unroll
  for (int t = 0; t < 16; t++) {
    int c = (t < 8) ? (kb + t) : (16 + kb + (t - 8));
    if (c < CHT)
      af[t] = (half_t)adjlist[(((long)b * CHT + c) << 16) + (i << 8) + j];
  }

  v8f acc[4];
#pragma unroll
  for (int nt = 0; nt < 4; nt++) acc[nt] = {};
#pragma unroll
  for (int nt = 0; nt < 4; nt++) {
    v16h bf = *(const v16h*)(Bfrag + ((long)nt * 32 + lane) * 16);
    acc[nt] = __builtin_amdgcn_wmma_f32_16x16x32_f16(false, af, false, bf, (short)0, acc[nt], false, false);
  }

  long rbase = eb + ((lane >= 16) ? 8 : 0);
#pragma unroll
  for (int nt = 0; nt < 4; nt++) {
    int bcol = nt * 16 + l15;
    float bv = bias[bcol];
#pragma unroll
    for (int rr = 0; rr < 8; rr++) {
      float v = acc[nt][rr] + bv;
      v = (v > 0.f) ? v : (__expf(v) - 1.f);
      Out[(rbase + rr) * 64 + bcol] = (half_t)v;
    }
  }
}

// ------------------------------ glue kernels -------------------------------
__global__ void f32_to_f16_k(const float* s, half_t* d, long n) {
  long t = (long)blockIdx.x * blockDim.x + threadIdx.x;
  if (t < n) d[t] = (half_t)s[t];
}

__global__ void pack_x_k(const float* x, half_t* d, int nodes, int cin) {
  int t = blockIdx.x * blockDim.x + threadIdx.x;
  if (t >= nodes * cin) return;
  int n = t / cin, k = t % cin;
  d[n * 32 + k] = (half_t)x[t];
}

__global__ void copy_ch0_k(const float* adj, float* adjlist) {
  long t = (long)blockIdx.x * blockDim.x + threadIdx.x;
  if (t >= (long)NB * NN2) return;
  int b = (int)(t >> 16); long r = t & (NN2 - 1);
  adjlist[((long)b * CHT) * NN2 + r] = adj[t];
}

// Pack an f32 weight matrix [rows x cols] into zero-padded WMMA B-fragment
// order for a KxNpad operand: dst[((nt*nch + kci)*32 + lane)*16 + tt].
__global__ void pack_wfrag_k(const float* W, half_t* dst, int rows, int cols,
                             int K, int Npad) {
  int nch = K >> 5;
  int n = (Npad >> 4) * nch * 512;
  int t = blockIdx.x * blockDim.x + threadIdx.x;
  if (t >= n) return;
  int tt = t & 15; int lane = (t >> 4) & 31; int rest = t >> 9;
  int kci = rest % nch; int nt = rest / nch;
  int l15 = lane & 15; int kb = (lane & 16) ? 8 : 0;
  int k = kci * 32 + ((tt < 8) ? (kb + tt) : (16 + kb + (tt - 8)));
  int c = nt * 16 + l15;
  dst[t] = (k < rows && c < cols) ? (half_t)W[k * cols + c] : (half_t)0.f;
}

__global__ void pack_b_k(const float* src, float* dst, int n, int np) {
  int t = blockIdx.x * blockDim.x + threadIdx.x;
  if (t < np) dst[t] = (t < n) ? src[t] : 0.f;
}

// degrees of (adj*(1-I)+I), clipped to >=1, ^-0.5
__global__ void deg_k(const float* adjlist, int ch_in, int C, float* dinv) {
  int t = blockIdx.x * blockDim.x + threadIdx.x;
  if (t >= NB * C * NN_) return;
  int b = t / (C * NN_); int rem = t % (C * NN_); int c = rem / NN_; int i = rem % NN_;
  const float* row = adjlist + ((long)b * CHT + ch_in + c) * NN2 + (long)i * NN_;
  float s = 0.f;
  for (int j = 0; j < NN_; j++) s += (j == i) ? 1.0f : row[j];
  dinv[t] = rsqrtf(fmaxf(s, 1.0f));
}

__global__ void adjn_k(const float* adjlist, int ch_in, int C, const float* dinv, half_t* adjn) {
  long t = (long)blockIdx.x * blockDim.x + threadIdx.x;
  long total = (long)NB * C * NN2; if (t >= total) return;
  int b = (int)(t / ((long)C * NN2)); long rem = t % ((long)C * NN2);
  int c = (int)(rem >> 16); int r = (int)(rem & (NN2 - 1)); int i = r >> 8, j = r & 255;
  float v = (i == j) ? 1.0f : adjlist[((long)b * CHT + ch_in + c) * NN2 + r];
  adjn[t] = (half_t)(dinv[(b * C + c) * NN_ + i] * v * dinv[(b * C + c) * NN_ + j]);
}

// xW_c = x @ W_c, written directly in WMMA B-fragment order per (b,c) batch:
// base = (b*C+c)*8192, frag dst = ((nt*8 + kci)*32 + lane)*16 + tt
__global__ void xw_k(const half_t* x16, const float* W, half_t* xw, int cin, int C, int c) {
  int t = blockIdx.x * blockDim.x + threadIdx.x;
  if (t >= 4096 * 32) return;
  int n = t >> 5, f = t & 31; int b = n >> 8, nl = n & 255;
  float s = 0.f;
  for (int k = 0; k < cin; k++) s += (float)x16[n * 32 + k] * W[k * 32 + f];
  int kci = nl >> 5; int kk = nl & 31;
  int lane = (f & 15) + ((kk & 8) ? 16 : 0);
  int tt = (kk & 7) + ((kk & 16) ? 8 : 0);
  int nt = f >> 4;
  xw[((long)b * C + c) * 8192 + (((long)nt * 8 + kci) * 32 + lane) * 16 + tt] = (half_t)s;
}

// concat channels for node MLP input
__global__ void mcpack_k(const half_t* h, half_t* mcA, int C) {
  int t = blockIdx.x * blockDim.x + threadIdx.x;
  int tot = 4096 * C * 32; if (t >= tot) return;
  int n = t / (C * 32); int k = t % (C * 32); int c = k >> 5, f = k & 31;
  int b = n >> 8, nl = n & 255;
  mcA[t] = h[(((long)b * C + c) * NN_ + nl) * 32 + f];
}

// a = a + a^T, masked by flags -> write into adj_list channel block
__global__ void sym_k(const float* raw, const float* flags, float* adjlist, int ch_out, int co) {
  long t = (long)blockIdx.x * blockDim.x + threadIdx.x;
  long tot = (long)NB * co * NN2; if (t >= tot) return;
  int b = (int)(t / ((long)co * NN2)); long rem = t % ((long)co * NN2);
  int c = (int)(rem >> 16); int r = (int)(rem & (NN2 - 1)); int i = r >> 8, j = r & 255;
  long rij = ((long)b << 16) + r;
  long rji = ((long)b << 16) + (j << 8) + i;
  float v = raw[rij * 16 + c] + raw[rji * 16 + c];
  v *= flags[(b << 8) + i] * flags[(b << 8) + j];
  adjlist[(((long)b * CHT + ch_out + c) << 16) + r] = v;
}

__global__ void score_k(const float* raw, const float* flags, float* out) {
  long t = (long)blockIdx.x * blockDim.x + threadIdx.x;
  if (t >= (long)NB * NN2) return;
  int b = (int)(t >> 16); int r = (int)(t & (NN2 - 1)); int i = r >> 8, j = r & 255;
  float v = raw[t * 16];
  if (i == j) v = 0.f;
  v *= flags[(b << 8) + i] * flags[(b << 8) + j];
  out[t] = v;
}

// ---------------------------------------------------------------------------
static inline dim3 gemm_grid(int M, int Np, int ntile, int batch) {
  long waves = (long)(M >> 4) * ((Np >> 4) / ntile);
  return dim3((unsigned)((waves + 7) / 8), (unsigned)batch);
}

extern "C" void kernel_launch(void* const* d_in, const int* in_sizes, int n_in,
                              void* d_out, int out_size, void* d_ws, size_t ws_size,
                              hipStream_t stream) {
  (void)n_in; (void)out_size; (void)ws_size;
  const float* x     = (const float*)d_in[0];
  const float* adj   = (const float*)d_in[1];
  const float* flags = (const float*)d_in[2];

  // ---- decode params (auto-detect layers-first vs final-first flatten) ----
  const int ci_arr[3]  = {2, 8, 8};
  const int co_arr[3]  = {8, 8, 4};
  const int chin[3]    = {0, 2, 10};
  const int chout[3]   = {2, 10, 18};
  const int cinf[3]    = {16, 32, 32};

  struct LayerP {
    const float* convW[8]; const float* convB[8];
    const float *mcW1, *mcB1, *mcW2, *mcB2;
    const float *mlW1, *mlB1, *mlW2, *mlB2;
  } LP[3];
  const float *fW1, *fB1, *fW2, *fB2;

  int idx = 3;
  auto nxt = [&]() -> const float* { return (const float*)d_in[idx++]; };
  auto readLayers = [&]() {
    for (int l = 0; l < 3; l++) {
      for (int c = 0; c < ci_arr[l]; c++) { LP[l].convW[c] = nxt(); LP[l].convB[c] = nxt(); }
      LP[l].mcW1 = nxt(); LP[l].mcB1 = nxt(); LP[l].mcW2 = nxt(); LP[l].mcB2 = nxt();
      LP[l].mlW1 = nxt(); LP[l].mlB1 = nxt(); LP[l].mlW2 = nxt(); LP[l].mlB2 = nxt();
    }
  };
  auto readFinal = [&]() { fW1 = nxt(); fB1 = nxt(); fW2 = nxt(); fB2 = nxt(); };
  if (in_sizes[3] == 22 * 44) { readFinal(); readLayers(); }
  else { readLayers(); readFinal(); }

  // ---- workspace arena ----
  char* base = (char*)d_ws;
  size_t off = 0;
  auto alloc = [&](size_t bytes) -> char* {
    char* p = base + off; off += (bytes + 255) & ~(size_t)255; return p;
  };
  float*  adjlist = (float*) alloc((size_t)NB * CHT * NN2 * 4); // all adj channels
  half_t* Hh      = (half_t*)alloc((size_t)ME * 64 * 2);        // hidden activations
  float*  raw     = (float*) alloc((size_t)ME * 16 * 4);        // edge/final raw f32 out
  half_t* adjn    = (half_t*)alloc((size_t)NB * 8 * NN2 * 2);   // normalized adj f16
  half_t* adjh    = (half_t*)alloc((size_t)NB * NN2 * 2);       // adj f16 for pow
  half_t* x16     = (half_t*)alloc((size_t)4096 * 32 * 2);      // node features f16
  half_t* xbuf    = (half_t*)alloc((size_t)4096 * 32 * 2);      // x_out f16
  half_t* xw      = (half_t*)alloc((size_t)NB * 8 * NN_ * 32 * 2); // frag-ordered
  half_t* hbuf    = (half_t*)alloc((size_t)NB * 8 * NN_ * 32 * 2);
  half_t* mcA     = (half_t*)alloc((size_t)4096 * 256 * 2);
  float*  dinv    = (float*) alloc((size_t)NB * 8 * NN_ * 4);

  half_t *wp_mc1[3], *wp_mc2[3], *wp_ml1[3], *wp_ml2[3], *wp_f1, *wp_f2;
  float  *bp_mc1[3], *bp_mc2[3], *bp_ml1[3], *bp_ml2[3], *convb[3], *bp_f1, *bp_f2;
  for (int l = 0; l < 3; l++) {
    int C = ci_arr[l];
    wp_mc1[l] = (half_t*)alloc((size_t)C * 32 * 32 * 2);
    wp_mc2[l] = (half_t*)alloc((size_t)32 * 32 * 2);
    wp_ml1[l] = (half_t*)alloc((size_t)96 * 32 * 2);
    wp_ml2[l] = (half_t*)alloc((size_t)32 * 16 * 2);
    bp_mc1[l] = (float*)alloc(32 * 4);
    bp_mc2[l] = (float*)alloc(32 * 4);
    bp_ml1[l] = (float*)alloc(32 * 4);
    bp_ml2[l] = (float*)alloc(16 * 4);
    convb[l]  = (float*)alloc((size_t)C * 32 * 4);
  }
  wp_f1 = (half_t*)alloc((size_t)32 * 64 * 2);
  wp_f2 = (half_t*)alloc((size_t)64 * 16 * 2);
  bp_f1 = (float*)alloc(64 * 4);
  bp_f2 = (float*)alloc(16 * 4);

  auto g1d = [](long n) { return dim3((unsigned)((n + 255) / 256)); };

  // ---- pack all weights (fragment order) / biases (zero-padded) ----
  for (int l = 0; l < 3; l++) {
    int C = ci_arr[l], co = co_arr[l];
    pack_wfrag_k<<<g1d(2 * C * 512), 256, 0, stream>>>(LP[l].mcW1, wp_mc1[l], C * 32, 8, C * 32, 32);
    pack_wfrag_k<<<g1d(2 * 1 * 512), 256, 0, stream>>>(LP[l].mcW2, wp_mc2[l], 8, 32, 32, 32);
    pack_wfrag_k<<<g1d(2 * 3 * 512), 256, 0, stream>>>(LP[l].mlW1, wp_ml1[l], 64 + C, 8, 96, 32);
    pack_wfrag_k<<<g1d(1 * 1 * 512), 256, 0, stream>>>(LP[l].mlW2, wp_ml2[l], 8, co, 32, 16);
    pack_b_k<<<g1d(32), 256, 0, stream>>>(LP[l].mcB1, bp_mc1[l], 8, 32);
    pack_b_k<<<g1d(32), 256, 0, stream>>>(LP[l].mcB2, bp_mc2[l], 32, 32);
    pack_b_k<<<g1d(32), 256, 0, stream>>>(LP[l].mlB1, bp_ml1[l], 8, 32);
    pack_b_k<<<g1d(16), 256, 0, stream>>>(LP[l].mlB2, bp_ml2[l], co, 16);
    for (int c = 0; c < C; c++)
      pack_b_k<<<g1d(32), 256, 0, stream>>>(LP[l].convB[c], convb[l] + c * 32, 32, 32);
  }
  pack_wfrag_k<<<g1d(4 * 1 * 512), 256, 0, stream>>>(fW1, wp_f1, 22, 44, 32, 64);
  pack_wfrag_k<<<g1d(1 * 2 * 512), 256, 0, stream>>>(fW2, wp_f2, 44, 1, 64, 16);
  pack_b_k<<<g1d(64), 256, 0, stream>>>(fB1, bp_f1, 44, 64);
  pack_b_k<<<g1d(16), 256, 0, stream>>>(fB2, bp_f2, 1, 16);

  // ---- pow tensor: channel0 = adj, channel1 = adj@adj (WMMA, row-major B) --
  copy_ch0_k<<<g1d((long)NB * NN2), 256, 0, stream>>>(adj, adjlist);
  f32_to_f16_k<<<g1d((long)NB * NN2), 256, 0, stream>>>(adj, adjh, (long)NB * NN2);
  gemm_k<0, 0, 1, 0><<<gemm_grid(NN_, NN_, 1, NB), 256, 0, stream>>>(
      adjh, NN_, NN2, adjh, NN_, NN2,
      (void*)(adjlist + NN2), NN_, (long)CHT * NN2,
      nullptr, 1, 0, nullptr, NN_, NN_, NN_);

  pack_x_k<<<g1d(4096 * 16), 256, 0, stream>>>(x, x16, 4096, 16);
  half_t* xcur = x16;

  // ---- layers ----
  for (int l = 0; l < 3; l++) {
    int C = ci_arr[l], co = co_arr[l], cin = cinf[l];
    deg_k<<<g1d((long)NB * C * NN_), 256, 0, stream>>>(adjlist, chin[l], C, dinv);
    adjn_k<<<g1d((long)NB * C * NN2), 256, 0, stream>>>(adjlist, chin[l], C, dinv, adjn);
    for (int c = 0; c < C; c++)
      xw_k<<<g1d(4096 * 32), 256, 0, stream>>>(xcur, LP[l].convW[c], xw, cin, C, c);
    // h[b,c] = adjn[b,c] @ xW[b,c] + conv_bias[c]   (batched WMMA, frag B)
    gemm_k<0, 1, 2, 1><<<gemm_grid(NN_, 32, 2, NB * C), 256, 0, stream>>>(
        adjn, NN_, NN2, xw, 0, 8192,
        hbuf, 32, (long)NN_ * 32, convb[l], C, 32, nullptr, NN_, NN_, 32);
    // node MLP: concat -> ELU -> tanh(* flags)
    mcpack_k<<<g1d((long)4096 * C * 32), 256, 0, stream>>>(hbuf, mcA, C);
    gemm_k<1, 1, 2, 1><<<gemm_grid(4096, 32, 2, 1), 256, 0, stream>>>(
        mcA, C * 32, 0, wp_mc1[l], 0, 0, Hh, 32, 0,
        bp_mc1[l], 1, 0, nullptr, 4096, C * 32, 32);
    gemm_k<2, 1, 2, 1><<<gemm_grid(4096, 32, 2, 1), 256, 0, stream>>>(
        Hh, 32, 0, wp_mc2[l], 0, 0, xbuf, 32, 0,
        bp_mc2[l], 1, 0, flags, 4096, 32, 32);
    // edge MLP over 1M rows: fused gather + WMMA, then second GEMM + sym
    edge_gemm1_k<<<g1d((long)(ME >> 4) * 32), 256, 0, stream>>>(
        xbuf, adjlist, chin[l], C, wp_ml1[l], bp_ml1[l], Hh);
    gemm_k<0, 0, 1, 1><<<gemm_grid(ME, 16, 1, 1), 256, 0, stream>>>(
        Hh, 32, 0, wp_ml2[l], 0, 0, raw, 16, 0,
        bp_ml2[l], 1, 0, nullptr, ME, 32, 16);
    sym_k<<<g1d((long)NB * co * NN2), 256, 0, stream>>>(raw, flags, adjlist, chout[l], co);
    xcur = xbuf;
  }

  // ---- final per-edge MLP 22 -> 44 -> 1 (fused gather layer-1) ----
  fin_gemm1_k<<<g1d((long)(ME >> 4) * 32), 256, 0, stream>>>(adjlist, wp_f1, bp_f1, Hh);
  gemm_k<0, 0, 1, 1><<<gemm_grid(ME, 16, 1, 1), 256, 0, stream>>>(
      Hh, 64, 0, wp_f2, 0, 0, raw, 16, 0, bp_f2, 1, 0, nullptr, ME, 64, 16);
  score_k<<<g1d((long)NB * NN2), 256, 0, stream>>>(raw, flags, (float*)d_out);
}
